// PointerNet_63548336112290
// MI455X (gfx1250) — matrix operand order repaired and checked by reading, not compiled, not run on hardware
//
#include <hip/hip_runtime.h>
#include <hip/hip_bf16.h>
#include <math.h>

// ---------------------------------------------------------------------------
// Pointer network (S=EMB=H=ATT=1024) as a single persistent gfx1250 kernel.
// Weights pre-converted to bf16 once; all mat-vec/GEMM on v_wmma_f32_16x16x32_bf16
// with b128 loads feeding the tiles. Bias pre-summed -> branchless f32x4 epilogue.
// ---------------------------------------------------------------------------

#define S_LEN 1024
#define EMB   1024
#define HID   1024
#define ATT   1024
#define G4    4096      // 4*HID
#define NBLK  32
#define NTHR  256       // 8 waves/block -> 256 waves total

typedef __bf16 bf16_t;
typedef bf16_t v16bf __attribute__((ext_vector_type(16)));
typedef float  v8f   __attribute__((ext_vector_type(8)));
typedef float  v4f   __attribute__((ext_vector_type(4)));

union BF16x16 {
    unsigned short s[16];
    uint4          q[2];
    v16bf          v;
};

__device__ __forceinline__ unsigned short f2bf(float f) {
    unsigned u = __float_as_uint(f);
    return (unsigned short)((u + 0x7FFFu + ((u >> 16) & 1u)) >> 16);  // RNE
}

__device__ __forceinline__ float sigmoidf_(float x) {
    return 1.0f / (1.0f + expf(-x));
}

// A tile (16x32 bf16) from row-major bf16 matrix. contiguous-K layout (same as B loaders):
// lane L: row m0+(L&15); halves[0..15] = K kc+off+0..15,
// off = (L>=16)?16:0.  Two global_load_b128 per tile.
__device__ __forceinline__ v16bf loadA_bf(const unsigned short* __restrict__ W, int ldw,
                                          int m0, int kc, int lane) {
    const unsigned short* p = W + (size_t)(m0 + (lane & 15)) * ldw + kc + ((lane >> 4) << 4);
    __builtin_prefetch(p + 64, 0, 3);   // global_prefetch_b8: next K chunks
    BF16x16 a;
    a.q[0] = *(const uint4*)p;
    a.q[1] = *(const uint4*)(p + 8);
    return a.v;
}

// Broadcast-vector B tile (32x16 bf16, identical columns) from LDS bf16 array.
// lanes 0-15 hold K=kc+0..15, lanes 16-31 hold K=kc+16..31. Two ds_load_b128.
__device__ __forceinline__ v16bf loadB_bcast(const unsigned short* lds, int kc, int lane) {
    const uint4* p = (const uint4*)(lds + kc + ((lane >> 4) << 4));
    BF16x16 b;
    b.q[0] = p[0];
    b.q[1] = p[1];
    return b.v;
}

// GEMM B tile: column n = row (n0+n) of row-major bf16 matrix (e = enc @ W1^T).
__device__ __forceinline__ v16bf loadB_rows_bf(const unsigned short* __restrict__ W, int ldw,
                                               int n0, int kc, int lane) {
    const unsigned short* p = W + (size_t)(n0 + (lane & 15)) * ldw + kc + ((lane >> 4) << 4);
    __builtin_prefetch(p + 64, 0, 3);
    BF16x16 b;
    b.q[0] = *(const uint4*)p;
    b.q[1] = *(const uint4*)(p + 8);
    return b.v;
}

// y[m0..m0+15] += W[m0..m0+15, 0..K) * x   (x staged bf16 in LDS, broadcast-B trick)
__device__ __forceinline__ v8f gemv_acc(v8f acc, const unsigned short* __restrict__ W,
                                        int ldw, int K, const unsigned short* xlds,
                                        int m0, int lane) {
#pragma unroll 2
    for (int kc = 0; kc < K; kc += 32) {
        v16bf a = loadA_bf(W, ldw, m0, kc, lane);
        v16bf b = loadB_bcast(xlds, kc, lane);
        acc = __builtin_amdgcn_wmma_f32_16x16x32_bf16(false, a, false, b,
                                                      (short)0, acc, false, false);
    }
    return acc;
}

// Column 0 of C lives in lanes 0 (M=0..7 in v[0..7]) and 16 (M=8..15).
// Branchless vectorized epilogue: bias pre-summed (b_ih+b_hh), f32x4 ops.
__device__ __forceinline__ void store_gemv_bias(v8f acc, float* __restrict__ out,
                                                const float* __restrict__ bsum,
                                                int m0, int lane) {
    if ((lane & 15) == 0) {
        const int mb = m0 + ((lane >> 4) << 3);          // multiple of 8 -> 32B aligned
        v4f b0 = *(const v4f*)(bsum + mb);
        v4f b1 = *(const v4f*)(bsum + mb + 4);
        v4f o0 = { acc[0] + b0[0], acc[1] + b0[1], acc[2] + b0[2], acc[3] + b0[3] };
        v4f o1 = { acc[4] + b1[0], acc[5] + b1[1], acc[6] + b1[2], acc[7] + b1[3] };
        *(v4f*)(out + mb)     = o0;
        *(v4f*)(out + mb + 4) = o1;
    }
}

__device__ __forceinline__ void store_gemv_nobias(v8f acc, float* __restrict__ out,
                                                  int m0, int lane) {
    if ((lane & 15) == 0) {
        const int mb = m0 + ((lane >> 4) << 3);
        v4f o0 = { acc[0], acc[1], acc[2], acc[3] };
        v4f o1 = { acc[4], acc[5], acc[6], acc[7] };
        *(v4f*)(out + mb)     = o0;
        *(v4f*)(out + mb + 4) = o1;
    }
}

// Agent-scope sense-reversing grid barrier.
__device__ __forceinline__ void gridbar(unsigned* cnt, unsigned* gen) {
    __syncthreads();
    if (threadIdx.x == 0) {
        unsigned g = __hip_atomic_load(gen, __ATOMIC_RELAXED, __HIP_MEMORY_SCOPE_AGENT);
        unsigned a = __hip_atomic_fetch_add(cnt, 1u, __ATOMIC_ACQ_REL, __HIP_MEMORY_SCOPE_AGENT);
        if (a == (unsigned)(NBLK - 1)) {
            __hip_atomic_store(cnt, 0u, __ATOMIC_RELAXED, __HIP_MEMORY_SCOPE_AGENT);
            __hip_atomic_fetch_add(gen, 1u, __ATOMIC_ACQ_REL, __HIP_MEMORY_SCOPE_AGENT);
        } else {
            while (__hip_atomic_load(gen, __ATOMIC_ACQUIRE, __HIP_MEMORY_SCOPE_AGENT) == g) {
                __builtin_amdgcn_s_sleep(2);
            }
        }
    }
    __syncthreads();
}

// One-time fp32 -> bf16 weight conversion (grid-strided).
__global__ void pn_cvt_kernel(const float* __restrict__ src,
                              unsigned short* __restrict__ dst, int n) {
    int i = blockIdx.x * blockDim.x + threadIdx.x;
    int stride = gridDim.x * blockDim.x;
    for (; i < n; i += stride) dst[i] = f2bf(src[i]);
}

// State init + pre-summed biases (bsum = b_ih + b_hh).
__global__ void pn_prep_kernel(float* h, float* c, float* maskv, unsigned* bar,
                               const float* __restrict__ ebih, const float* __restrict__ ebhh,
                               const float* __restrict__ dbih, const float* __restrict__ dbhh,
                               float* __restrict__ ebsum, float* __restrict__ dbsum) {
    int i = blockIdx.x * blockDim.x + threadIdx.x;
    if (i < HID) { h[i] = 0.f; c[i] = 0.f; }
    if (i < S_LEN) maskv[i] = 1.f;
    if (i < G4) {
        ebsum[i] = ebih[i] + ebhh[i];
        dbsum[i] = dbih[i] + dbhh[i];
    }
    if (i == 0) { bar[0] = 0u; bar[1] = 0u; }
}

__global__ __launch_bounds__(NTHR, 1) void pn_main_kernel(
    const float* __restrict__ seq, const int* __restrict__ order,
    const float* __restrict__ d0,
    const unsigned short* __restrict__ eWih, const unsigned short* __restrict__ eWhh,
    const unsigned short* __restrict__ dWih, const unsigned short* __restrict__ dWhh,
    const unsigned short* __restrict__ W1, const unsigned short* __restrict__ W2,
    const float* __restrict__ ebsum, const float* __restrict__ dbsum,
    const float* __restrict__ Vv,
    float* __restrict__ out,          // [1025] : pred_order (as float) + loss
    unsigned* bar,                    // [2] barrier cnt/gen
    float* h, float* c,               // [HID] recurrent state (enc then dec)
    float* gbuf,                      // [4096] raw gates
    float* dvec,                      // [ATT]  W2 @ hd
    float* uvec,                      // [S]    scores
    float* avec,                      // [S]    softmax
    float* divec,                     // [HID]  attention readout
    float* maskv,                     // [S]
    float* enc_out,                   // [S*HID] fp32
    unsigned short* enc_bf,           // [S*HID] bf16
    float* emat)                      // [S*ATT] e = enc_out @ W1^T
{
    __shared__ __align__(16) unsigned short sx[EMB + HID];  // staged input (bf16)
    __shared__ __align__(16) unsigned short sh[HID];        // staged hidden (bf16)
    __shared__ float red[NTHR];
    __shared__ int   redi[NTHR];
    __shared__ float s_gm, s_sum;
    __shared__ int   s_gi;

    const int tid  = threadIdx.x;
    const int lane = tid & 31;
    const int gid  = blockIdx.x * NTHR + tid;
    const int wv   = gid >> 5;           // global wave id, 0..255
    float loss_local = 0.f;              // meaningful only in block0/thread0

    // ----------------- encoder LSTM: 1024 sequential steps -----------------
#pragma unroll 1
    for (int t = 0; t < S_LEN; ++t) {
        for (int j = tid; j < HID; j += NTHR) {
            sx[j] = f2bf(seq[(size_t)t * EMB + j]);
            sh[j] = f2bf(h[j]);
        }
        __syncthreads();

        const int m0 = wv * 16;                     // 256 waves cover 4096 rows
        v8f acc = {};
        acc = gemv_acc(acc, eWih, EMB, EMB, sx, m0, lane);
        acc = gemv_acc(acc, eWhh, HID, HID, sh, m0, lane);
        store_gemv_bias(acc, gbuf, ebsum, m0, lane);
        gridbar(bar, bar + 1);

        if (gid < HID) {
            float ig = sigmoidf_(gbuf[gid]);
            float fg = sigmoidf_(gbuf[HID + gid]);
            float gg = tanhf(gbuf[2 * HID + gid]);
            float og = sigmoidf_(gbuf[3 * HID + gid]);
            float cn = fg * c[gid] + ig * gg;
            float hn = og * tanhf(cn);
            c[gid] = cn;
            h[gid] = hn;
            enc_out[(size_t)t * HID + gid] = hn;
            enc_bf[(size_t)t * HID + gid] = f2bf(hn);
        }
        gridbar(bar, bar + 1);
    }

    // ----------------- e = enc_out @ W1^T : 1024x1024x1024 WMMA GEMM --------
#pragma unroll 1
    for (int it = 0; it < 16; ++it) {
        const int T  = wv * 16 + it;                // 4096 output tiles
        const int s0 = (T >> 6) << 4;
        const int a0 = (T & 63) << 4;
        v8f acc = {};
#pragma unroll 2
        for (int kc = 0; kc < HID; kc += 32) {
            v16bf a = loadA_bf(enc_bf, HID, s0, kc, lane);
            v16bf b = loadB_rows_bf(W1, HID, a0, kc, lane);
            acc = __builtin_amdgcn_wmma_f32_16x16x32_bf16(false, a, false, b,
                                                          (short)0, acc, false, false);
        }
        const int rb  = s0 + ((lane >> 4) << 3);
        const int col = a0 + (lane & 15);
#pragma unroll
        for (int r = 0; r < 8; ++r)
            emat[(size_t)(rb + r) * ATT + col] = acc[r];
    }
    gridbar(bar, bar + 1);

    // ----------------- pointer decoder: 1024 sequential steps ---------------
#pragma unroll 1
    for (int t = 0; t < S_LEN; ++t) {
        for (int j = tid; j < HID; j += NTHR) sh[j] = f2bf(h[j]);
        __syncthreads();

        // d = W2 @ hd   (waves 0..63)
        if (wv < 64) {
            v8f acc = {};
            acc = gemv_acc(acc, W2, HID, HID, sh, wv * 16, lane);
            store_gemv_nobias(acc, dvec, wv * 16, lane);
        }
        gridbar(bar, bar + 1);

        // u[s] = V . tanh(d + e[s,:]) ; masked   (one wave per 4 rows)
#pragma unroll 1
        for (int q = 0; q < 4; ++q) {
            const int s = wv * 4 + q;
            const float* er = emat + (size_t)s * ATT;
            float sum = 0.f;
            for (int a_ = lane; a_ < ATT; a_ += 32)
                sum += tanhf(dvec[a_] + er[a_]) * Vv[a_];
#pragma unroll
            for (int off = 16; off; off >>= 1) sum += __shfl_xor(sum, off, 32);
            if (lane == 0) uvec[s] = (maskv[s] == 0.f) ? -INFINITY : sum;
        }
        gridbar(bar, bar + 1);

        // block 0: softmax / argmax / loss / mask update
        if (blockIdx.x == 0) {
            float m = -INFINITY; int mi = 1 << 20;
            for (int s = tid; s < S_LEN; s += NTHR) {
                float v = uvec[s];
                if (v > m || (v == m && s < mi)) { m = v; mi = s; }
            }
            red[tid] = m; redi[tid] = mi;
            __syncthreads();
            if (tid == 0) {
                float gm = -INFINITY; int gi = 1 << 20;
                for (int i = 0; i < NTHR; ++i)
                    if (red[i] > gm || (red[i] == gm && redi[i] < gi)) { gm = red[i]; gi = redi[i]; }
                s_gm = gm; s_gi = gi;
            }
            __syncthreads();
            const float gm = s_gm;
            float ps = 0.f;
            for (int s = tid; s < S_LEN; s += NTHR) ps += expf(uvec[s] - gm);
            __syncthreads();
            red[tid] = ps;
            __syncthreads();
            if (tid == 0) {
                float ss = 0.f;
                for (int i = 0; i < NTHR; ++i) ss += red[i];
                s_sum = ss;
            }
            __syncthreads();
            const float gs = s_sum;
            for (int s = tid; s < S_LEN; s += NTHR)
                avec[s] = expf(uvec[s] - gm) / gs;
            if (tid == 0) {
                const int ti = order[t];
                loss_local += -(uvec[ti] - gm - logf(gs));
                out[t] = (float)s_gi;               // predicted index
                maskv[ti] = 0.f;                    // visit true_i
            }
        }
        gridbar(bar, bar + 1);

        // di = a @ enc_out (coalesced column reduction)
        if (gid < HID) {
            float acc = 0.f;
            for (int s = 0; s < S_LEN; ++s)
                acc += avec[s] * enc_out[(size_t)s * HID + gid];
            divec[gid] = acc;
        }
        gridbar(bar, bar + 1);

        // decoder LSTM: x = [di, dec_in], gates via WMMA
        for (int j = tid; j < HID; j += NTHR) {
            sx[j] = f2bf(divec[j]);
            float xin = (t == 0) ? d0[j] : seq[(size_t)order[t - 1] * EMB + j];
            sx[HID + j] = f2bf(xin);
        }
        __syncthreads();
        {
            const int m0 = wv * 16;
            v8f acc = {};
            acc = gemv_acc(acc, dWih, EMB + HID, EMB + HID, sx, m0, lane);
            acc = gemv_acc(acc, dWhh, HID, HID, sh, m0, lane);
            store_gemv_bias(acc, gbuf, dbsum, m0, lane);
        }
        gridbar(bar, bar + 1);

        if (gid < HID) {
            float ig = sigmoidf_(gbuf[gid]);
            float fg = sigmoidf_(gbuf[HID + gid]);
            float gg = tanhf(gbuf[2 * HID + gid]);
            float og = sigmoidf_(gbuf[3 * HID + gid]);
            float cn = fg * c[gid] + ig * gg;
            float hn = og * tanhf(cn);
            c[gid] = cn;
            h[gid] = hn;
        }
        gridbar(bar, bar + 1);
    }

    if (blockIdx.x == 0 && tid == 0) out[S_LEN] = loss_local;
}

extern "C" void kernel_launch(void* const* d_in, const int* in_sizes, int n_in,
                              void* d_out, int out_size, void* d_ws, size_t ws_size,
                              hipStream_t stream) {
    (void)in_sizes; (void)n_in; (void)out_size; (void)ws_size;

    const float* seq   = (const float*)d_in[0];
    const int*   order = (const int*)  d_in[1];
    const float* d0    = (const float*)d_in[2];
    const float* eWih  = (const float*)d_in[3];
    const float* eWhh  = (const float*)d_in[4];
    const float* ebih  = (const float*)d_in[5];
    const float* ebhh  = (const float*)d_in[6];
    const float* dWih  = (const float*)d_in[7];
    const float* dWhh  = (const float*)d_in[8];
    const float* dbih  = (const float*)d_in[9];
    const float* dbhh  = (const float*)d_in[10];
    const float* W1    = (const float*)d_in[11];
    const float* W2    = (const float*)d_in[12];
    const float* Vv    = (const float*)d_in[13];
    float* out = (float*)d_out;

    // workspace carve-up (256B aligned slices)
    char* w = (char*)d_ws;
    auto take = [&](size_t bytes) { char* p = w; w += (bytes + 255) & ~(size_t)255; return p; };
    unsigned*       bar      = (unsigned*)      take(256);
    float*          h        = (float*)         take(HID * 4);
    float*          c        = (float*)         take(HID * 4);
    float*          gbuf     = (float*)         take(G4 * 4);
    float*          dvec     = (float*)         take(ATT * 4);
    float*          uvec     = (float*)         take(S_LEN * 4);
    float*          avec     = (float*)         take(S_LEN * 4);
    float*          divec    = (float*)         take(HID * 4);
    float*          maskv    = (float*)         take(S_LEN * 4);
    float*          ebsum    = (float*)         take(G4 * 4);
    float*          dbsum    = (float*)         take(G4 * 4);
    float*          enc_out  = (float*)         take((size_t)S_LEN * HID * 4);
    unsigned short* enc_bf   = (unsigned short*)take((size_t)S_LEN * HID * 2);
    float*          emat     = (float*)         take((size_t)S_LEN * ATT * 4);
    // bf16 weight copies (converted once per call)
    unsigned short* eWih_bf  = (unsigned short*)take((size_t)G4 * EMB * 2);
    unsigned short* eWhh_bf  = (unsigned short*)take((size_t)G4 * HID * 2);
    unsigned short* dWih_bf  = (unsigned short*)take((size_t)G4 * (EMB + HID) * 2);
    unsigned short* dWhh_bf  = (unsigned short*)take((size_t)G4 * HID * 2);
    unsigned short* W1_bf    = (unsigned short*)take((size_t)ATT * HID * 2);
    unsigned short* W2_bf    = (unsigned short*)take((size_t)ATT * HID * 2);

    const int cvtBlk = 1024, cvtThr = 256;
    pn_cvt_kernel<<<cvtBlk, cvtThr, 0, stream>>>(eWih, eWih_bf, G4 * EMB);
    pn_cvt_kernel<<<cvtBlk, cvtThr, 0, stream>>>(eWhh, eWhh_bf, G4 * HID);
    pn_cvt_kernel<<<cvtBlk, cvtThr, 0, stream>>>(dWih, dWih_bf, G4 * (EMB + HID));
    pn_cvt_kernel<<<cvtBlk, cvtThr, 0, stream>>>(dWhh, dWhh_bf, G4 * HID);
    pn_cvt_kernel<<<cvtBlk, cvtThr, 0, stream>>>(W1, W1_bf, ATT * HID);
    pn_cvt_kernel<<<cvtBlk, cvtThr, 0, stream>>>(W2, W2_bf, ATT * HID);
    pn_prep_kernel<<<16, 256, 0, stream>>>(h, c, maskv, bar,
                                           ebih, ebhh, dbih, dbhh, ebsum, dbsum);

    pn_main_kernel<<<NBLK, NTHR, 0, stream>>>(
        seq, order, d0, eWih_bf, eWhh_bf, dWih_bf, dWhh_bf, W1_bf, W2_bf,
        ebsum, dbsum, Vv, out, bar, h, c, gbuf, dvec, uvec, avec, divec, maskv,
        enc_out, enc_bf, emat);
}
